// ConvShapeletNet_35072702939461
// MI455X (gfx1250) — matrix operand order repaired and checked by reading, not compiled, not run on hardware
//
#include <hip/hip_runtime.h>
#include <hip/hip_bf16.h>

// Problem constants (from reference)
constexpr int B_      = 512;
constexpr int L_      = 8192;
constexpr int C_      = 10;
constexpr int L_OUT   = 25;            // windows per (b,c)
constexpr int L_P     = 8;             // pooled features per (b,c)
constexpr int NFEAT   = 80;            // 10 * 8
constexpr int NCLS    = 10;
constexpr int KPAD    = 96;            // K=80 padded to 3 WMMA steps of 32

// gcd(BAG=1146, STRIDE=286) == 2 -> pair-sum granularity makes every window an
// exact span of pair sums: window j = pairsum[143*j .. 143*j + 573)
constexpr int NPAIR   = L_ / 2;        // 4096 pair sums per (b,c) chunk
constexpr int WSTART  = 143;
constexpr int WLEN    = 573;

typedef __attribute__((ext_vector_type(16))) _Float16 v16h;
typedef __attribute__((ext_vector_type(8)))  float    v8f;

// ---------------------------------------------------------------------------
// Kernel 1: bandwidth kernel. One workgroup per (b,c) chunk of 8192 floats.
// Reads 32KB coalesced (each global byte touched exactly once), produces 8
// pooled features. ~19 blocks fit per 320KB WGP -> plenty of latency hiding.
// ---------------------------------------------------------------------------
__global__ __launch_bounds__(256) void winpool_kernel(
    const float* __restrict__ x,          // [B, L, C] flat == B*C chunks of 8192
    const float* __restrict__ conv_bias,  // [10]
    float* __restrict__ feat)             // [B, 80]
{
    __shared__ float ps[NPAIR];           // 16 KB pair sums
    __shared__ float win[32];             // squared window values (25 used)

    const int bc   = blockIdx.x;          // 0 .. B*C-1
    const int b    = bc / C_;
    const int c    = bc % C_;
    const int tid  = threadIdx.x;

    // Stage 1: coalesced float2 loads -> pair sums in LDS.
    const float2* xp = reinterpret_cast<const float2*>(x) + (size_t)bc * (size_t)NPAIR;
#pragma unroll
    for (int i = 0; i < NPAIR / 256; ++i) {
        int k = i * 256 + tid;
        float2 v = xp[k];
        ps[k] = v.x + v.y;
    }
    __syncthreads();

    // Stage 2: each of the 8 waves reduces windows j = wave, wave+8, wave+16.
    const int wave = tid >> 5;
    const int lane = tid & 31;
    const float cb = conv_bias[c];
    for (int j = wave; j < L_OUT; j += 8) {
        float s = 0.0f;
        const int base = WSTART * j;
        for (int idx = lane; idx < WLEN; idx += 32)   // stride-32: bank-conflict free
            s += ps[base + idx];
        for (int off = 16; off > 0; off >>= 1)        // wave32 reduction
            s += __shfl_xor(s, off, 32);
        if (lane == 0) {
            float t = s + cb;
            win[j] = t * t;
        }
    }
    __syncthreads();

    // Stage 3: maxpool(-y, 3) == -(min over 3 of y). First 24 windows used.
    if (tid < L_P) {
        float m = fminf(fminf(win[3 * tid], win[3 * tid + 1]), win[3 * tid + 2]);
        feat[(size_t)b * NFEAT + c * L_P + tid] = -m;
    }
}

// ---------------------------------------------------------------------------
// Kernel 2: FC out[512,10] = feat[512,80] @ fc_w[10,80]^T + fc_b
// via v_wmma_f32_16x16x32_f16. One wave per 16-row M tile (32 blocks).
// Operands are staged through zero-padded LDS tiles so fragment construction
// is branchless (no divergent guarded loads) and K/N padding is free.
// ---------------------------------------------------------------------------
__global__ __launch_bounds__(32) void fc_wmma_kernel(
    const float* __restrict__ feat,    // [512, 80]
    const float* __restrict__ fc_w,    // [10, 80]
    const float* __restrict__ fc_b,    // [10]
    float* __restrict__ out)           // [512, 10]
{
    __shared__ _Float16 As[16 * KPAD]; // 16 rows (M) x 96 (K padded), 3 KB
    __shared__ _Float16 Bs[16 * KPAD]; // 16 rows (N padded) x 96 (K padded), 3 KB

    const int tile = blockIdx.x;       // 0..31 (M tiles of 16)
    const int lane = threadIdx.x;      // 0..31
    const int hi   = lane >> 4;        // lane group
    const int lo   = lane & 15;        // A: M row within tile; B/D: N column

    // Zero both staging tiles (covers K 80..95 pad and B rows 10..15).
    for (int i = lane; i < 16 * KPAD; i += 32) {
        As[i] = (_Float16)0.0f;
        Bs[i] = (_Float16)0.0f;
    }
    // Fill valid region, converting f32 -> f16. Coalesced along k.
    for (int i = lane; i < 16 * NFEAT; i += 32) {
        int r = i / NFEAT, k = i % NFEAT;
        As[r * KPAD + k] = (_Float16)feat[(size_t)(tile * 16 + r) * NFEAT + k];
    }
    for (int i = lane; i < NCLS * NFEAT; i += 32) {
        int r = i / NFEAT, k = i % NFEAT;
        Bs[r * KPAD + k] = (_Float16)fc_w[r * NFEAT + k];
    }
    __syncthreads();

    v8f acc = {};
#pragma unroll
    for (int step = 0; step < 3; ++step) {
        const int kbase = step * 32;
        v16h a, bm;
#pragma unroll
        for (int h = 0; h < 16; ++h) {
            // A fragment (16-bit A 16x32): lane = M, K = (h<8?h:h+8) + 8*grp.
            a[h]  = As[lo * KPAD + kbase + (h < 8 ? h : h + 8) + hi * 8];
            // B fragment (32x16): lane = N, K = (h<8?h:h+8) + 8*grp.
            bm[h] = Bs[lo * KPAD + kbase + (h < 8 ? h : h + 8) + hi * 8];
        }
        // D = A*B + C, f32 accumulate. Full-wave EXEC, no divergence here.
        acc = __builtin_amdgcn_wmma_f32_16x16x32_f16(
            /*neg_a=*/false, a, /*neg_b=*/false, bm,
            /*c_mod=*/(short)0, acc, /*reuse_a=*/false, /*reuse_b=*/false);
    }

    // D layout: VGPR r, lanes 0-15 -> (M=r, N=lane); lanes 16-31 -> (M=r+8, N=lane-16)
    const int n = lo;
    if (n < NCLS) {
        const float bias = fc_b[n];
#pragma unroll
        for (int r = 0; r < 8; ++r) {
            const int m = tile * 16 + r + 8 * hi;
            out[m * NCLS + n] = acc[r] + bias;
        }
    }
}

// ---------------------------------------------------------------------------
extern "C" void kernel_launch(void* const* d_in, const int* in_sizes, int n_in,
                              void* d_out, int out_size, void* d_ws, size_t ws_size,
                              hipStream_t stream) {
    const float* x         = (const float*)d_in[0];   // [512*8192*10]
    const float* conv_bias = (const float*)d_in[1];   // [10]
    const float* fc_w      = (const float*)d_in[2];   // [10*80]
    const float* fc_b      = (const float*)d_in[3];   // [10]
    float*       out       = (float*)d_out;           // [512*10]
    float*       feat      = (float*)d_ws;            // [512*80] scratch

    winpool_kernel<<<B_ * C_, 256, 0, stream>>>(x, conv_bias, feat);
    fc_wmma_kernel<<<B_ / 16, 32, 0, stream>>>(feat, fc_w, fc_b, out);
}
